// mrg_26877905339009
// MI455X (gfx1250) — matrix operand order repaired and checked by reading, not compiled, not run on hardware
//
#include <hip/hip_runtime.h>
#include <hip/hip_bf16.h>

typedef unsigned short u16;
typedef __attribute__((ext_vector_type(16))) __bf16 v16bf;
typedef __attribute__((ext_vector_type(8)))  float  v8f;

#define T_STEPS 50
#define BATCH   256
#define TB      (T_STEPS*BATCH)   // 12800
#define HID     512
#define G3      1536              // 3*H
#define GIN     576               // WD + MLP1
#define VOC     20000
#define WDM     512
#define MLP0D   128
#define MLP1D   64

__device__ __forceinline__ u16 f2bf(float x) {
  __hip_bfloat16 h = __float2bfloat16(x);
  union { __hip_bfloat16 h; u16 u; } c; c.h = h; return c.u;
}

union BFrag { uint4 q[2]; v16bf v; };

__device__ __forceinline__ v8f wmma_bf16(v16bf a, v16bf b, v8f c) {
  return __builtin_amdgcn_wmma_f32_16x16x32_bf16(false, a, false, b, (short)0, c,
                                                 false, false);
}

// A fragment: 16(M)x32(K) bf16 from row-major [rows, stride], tile at (row0, k0).
// ISA 7.12.2: lane<16 -> M=lane, K {k0..k0+7} in v0-3, {k0+16..k0+23} in v4-7;
//             lane>=16 -> M=lane-16, K {k0+8..15}, {k0+24..31}.
__device__ __forceinline__ v16bf load_a_frag(const u16* base, int stride, int row0,
                                             int k0, int lane) {
  int m  = lane & 15;
  int kb = k0 + ((lane >> 4) << 3);
  const u16* p = base + (size_t)(row0 + m) * stride + kb;
  BFrag f;
  f.q[0] = *(const uint4*)(p);
  f.q[1] = *(const uint4*)(p + 16);
  return f.v;
}

// B fragment: 32(K)x16(N) bf16; source stored row-major [N, stride] so that
// output column n is contiguous in K (w_ih/w_hh native layout; out_W pre-transposed).
// lanes 0-15: N=lane, K {k0..k0+7},{k0+16..k0+23}; lanes 16-31: N=lane-16, K {k0+8..15},{k0+24..31}.
__device__ __forceinline__ v16bf load_b_frag(const u16* base, int stride, int n0,
                                             int k0, int lane) {
  int n  = n0 + (lane & 15);
  int kb = k0 + ((lane >> 4) << 3);
  const u16* p = base + (size_t)n * stride + kb;
  BFrag f;
  f.q[0] = *(const uint4*)(p);
  f.q[1] = *(const uint4*)(p + 16);
  return f.v;
}

// ---------------- encoder (tiny, fp32 VALU) ----------------
__global__ void enc1_kernel(const int* user, const int* item, const float* uemb,
                            const float* iemb, const float* w0, float* m0) {
  __shared__ float cat[128];
  int b = blockIdx.x, tid = threadIdx.x;  // 128 threads
  if (tid < 64) cat[tid] = uemb[(size_t)user[b] * 64 + tid];
  else          cat[tid] = iemb[(size_t)item[b] * 64 + (tid - 64)];
  __syncthreads();
  float acc = 0.f;
  for (int i = 0; i < 128; ++i) acc += cat[i] * w0[i * 128 + tid];
  m0[b * 128 + tid] = tanhf(acc);
}

__global__ void enc2_kernel(const float* m0, const float* w1, float* m) {
  __shared__ float in[128];
  int b = blockIdx.x, tid = threadIdx.x;  // 64 threads
  in[tid] = m0[b * 128 + tid];
  in[tid + 64] = m0[b * 128 + tid + 64];
  __syncthreads();
  float acc = 0.f;
  for (int i = 0; i < 128; ++i) acc += in[i] * w1[i * 64 + tid];
  m[b * 64 + tid] = tanhf(acc);
}

__global__ void enc3_kernel(const float* m, const float* hW, const float* hbias,
                            float* h0) {
  __shared__ float in[64];
  int b = blockIdx.x, tid = threadIdx.x;  // 256 threads
  if (tid < 64) in[tid] = m[b * 64 + tid];
  __syncthreads();
  for (int c = tid; c < HID; c += 256) {
    float acc = hbias[c];
    for (int i = 0; i < 64; ++i) acc += in[i] * hW[i * HID + c];
    h0[b * HID + c] = tanhf(acc);
  }
}

__global__ void rating_kernel(const float* m, const float* fW, const float* fb,
                              const int* user, const int* item, const float* ubias,
                              const float* ibias, const float* label,
                              float* out_rating, float* rating_sum) {
  __shared__ float red[256];
  int b = threadIdx.x;  // 1 block, 256 threads
  float acc = fb[0];
  for (int i = 0; i < 64; ++i) acc += m[b * 64 + i] * fW[i];
  float pred = acc + 3.5f + ubias[user[b]] + ibias[item[b]];
  float d = pred - label[b];
  float d2 = d * d;
  out_rating[b] = d2;
  red[b] = d2;
  __syncthreads();
  for (int s = 128; s > 0; s >>= 1) {
    if (b < s) red[b] += red[b + s];
    __syncthreads();
  }
  if (b == 0) *rating_sum = red[0];
}

// ---------------- bf16 prep ----------------
__global__ void zero_scal_kernel(float* s) {
  if (threadIdx.x < 4) s[threadIdx.x] = 0.f;
}

__global__ void cvt_bf16_kernel(const float* src, u16* dst, int n) {
  int i = blockIdx.x * 256 + threadIdx.x;
  if (i < n) dst[i] = f2bf(src[i]);
}

// out_W [HID, VOC] -> owt [VOC, HID] bf16
__global__ void transpose_outw_kernel(const float* w, u16* wt) {
  int i = blockIdx.x * 256 + threadIdx.x;
  if (i < VOC * HID) {
    int v = i >> 9, h = i & 511;
    wt[i] = f2bf(w[(size_t)h * VOC + v]);
  }
}

// x[row, 0:512] = word_emb[review_input[row]]; x[row, 512:576] = m[row%256]
__global__ void build_x_kernel(const int* rin, const float* wemb, const float* m,
                               u16* xb) {
  int i = blockIdx.x * 256 + threadIdx.x;
  if (i < TB * GIN) {
    int row = i / GIN, c = i - row * GIN;
    float v = (c < WDM) ? wemb[(size_t)rin[row] * WDM + c]
                        : m[(row & 255) * 64 + (c - WDM)];
    xb[i] = f2bf(v);
  }
}

// ---------------- fused GRU (gx + gh WMMA + gates) ----------------
// grid 16 blocks x 256 threads; block owns 16 batch rows across all T steps.
// Dynamic LDS: gxs[16][1536] f32, ghs[16][1536] f32, hf[16][512] f32, hb[16][512] bf16.
__global__ void gru_kernel(const u16* __restrict__ xb,   // [TB, 576] bf16
                           const u16* __restrict__ wih,  // [1536, 576] bf16
                           const u16* __restrict__ whh,  // [1536, 512] bf16
                           const float* bih, const float* bhh,
                           const float* h0,              // [256, 512] f32
                           u16* __restrict__ outs) {     // [TB, 512] bf16
  extern __shared__ char smem[];
  float* gxs = (float*)smem;                    // 16*1536
  float* ghs = gxs + 16 * G3;                   // 16*1536
  float* hf  = ghs + 16 * G3;                   // 16*512
  u16*   hb  = (u16*)(hf + 16 * HID);           // 16*512

  const int b0 = blockIdx.x * 16;
  const int tid = threadIdx.x;
  const int lane = tid & 31, wave = tid >> 5;

  for (int i = tid; i < 16 * HID; i += 256) {
    float v = h0[(size_t)(b0 + (i >> 9)) * HID + (i & 511)];
    hf[i] = v;
    hb[i] = f2bf(v);
  }
  __syncthreads();

  for (int t = 0; t < T_STEPS; ++t) {
    const u16* xrow = xb + (size_t)(t * BATCH + b0) * GIN;
    // 96 gx tiles + 96 gh tiles over 8 waves
    for (int job = wave; job < 192; job += 8) {
      const bool is_gx = (job < 96);
      const int nt = is_gx ? job : job - 96;
      const int g0 = nt * 16;
      v8f acc = {};
      if (is_gx) {
        for (int k = 0; k < GIN / 32; ++k) {
          v16bf a = load_a_frag(xrow, GIN, 0, k * 32, lane);
          v16bf b = load_b_frag(wih, GIN, g0, k * 32, lane);
          acc = wmma_bf16(a, b, acc);
        }
      } else {
        for (int k = 0; k < HID / 32; ++k) {
          v16bf a = load_a_frag(hb, HID, 0, k * 32, lane);  // LDS source
          v16bf b = load_b_frag(whh, HID, g0, k * 32, lane);
          acc = wmma_bf16(a, b, acc);
        }
      }
      const float bias = is_gx ? bih[g0 + (lane & 15)] : bhh[g0 + (lane & 15)];
      float* dst = is_gx ? gxs : ghs;
      const int col = g0 + (lane & 15);
      const int rbase = (lane >> 4) << 3;
#pragma unroll
      for (int r = 0; r < 8; ++r)
        dst[(rbase + r) * G3 + col] = acc[r] + bias;
    }
    __syncthreads();
    // gates
    for (int i = tid; i < 16 * HID; i += 256) {
      int b = i >> 9, j = i & 511;
      float xr = gxs[b * G3 + j], xz = gxs[b * G3 + 512 + j],
            xn = gxs[b * G3 + 1024 + j];
      float hr = ghs[b * G3 + j], hz = ghs[b * G3 + 512 + j],
            hn = ghs[b * G3 + 1024 + j];
      float r = 1.f / (1.f + __expf(-(xr + hr)));
      float z = 1.f / (1.f + __expf(-(xz + hz)));
      float n = tanhf(xn + r * hn);
      float h = (1.f - z) * n + z * hf[i];
      hf[i] = h;
      u16 hv = f2bf(h);
      hb[i] = hv;
      outs[(size_t)(t * BATCH + b0 + b) * HID + j] = hv;
    }
    __syncthreads();
  }
}

// ---------------- vocab GEMM + online log-softmax + CE ----------------
// grid 200 blocks x 256 threads; block owns 64 rows (4 M-tiles). Each wave
// processes 2 N-tiles per iteration so A-fragments are reused across two
// B-tiles (1.5 b128 loads per WMMA instead of 2.5). Logits never hit memory.
__global__ void logits_ce_kernel(const u16* __restrict__ outs,  // [TB, 512] bf16
                                 const u16* __restrict__ owt,   // [VOC, 512] bf16
                                 const float* outb,             // [VOC]
                                 const int* tgt,                // [TB]
                                 float* out_review,             // [TB]
                                 float* review_sum, float* mask_sum) {
  __shared__ float ls_M[8 * 64];
  __shared__ float ls_S[8 * 64];
  __shared__ float ls_T[64];
  __shared__ int ls_tgt[64];
  const int rowBase = blockIdx.x * 64;
  const int tid = threadIdx.x, lane = tid & 31, wave = tid >> 5;

  for (int i = tid; i < 8 * 64; i += 256) { ls_M[i] = -3.0e38f; ls_S[i] = 0.f; }
  for (int i = tid; i < 64; i += 256) { ls_tgt[i] = tgt[rowBase + i]; ls_T[i] = 0.f; }
  __syncthreads();

  // 1250 N-tiles, 8 waves x 2 tiles per iteration -> stride 16, no remainder.
  for (int nt0 = wave * 2; nt0 < VOC / 16; nt0 += 16) {
    const int n0 = nt0 * 16;  // this iteration covers columns n0 .. n0+31
    // prefetch next iteration's B rows (gfx1250 global_prefetch_b8)
    {
      int pn = n0 + 256 + lane;
      if (pn < VOC) __builtin_prefetch(owt + (size_t)pn * HID, 0, 1);
    }
    v8f acc[2][4] = {{v8f{}, v8f{}, v8f{}, v8f{}}, {v8f{}, v8f{}, v8f{}, v8f{}}};
    for (int k = 0; k < HID / 32; ++k) {
      v16bf bf0 = load_b_frag(owt, HID, n0, k * 32, lane);
      v16bf bf1 = load_b_frag(owt, HID, n0 + 16, k * 32, lane);
#pragma unroll
      for (int mt = 0; mt < 4; ++mt) {
        v16bf a = load_a_frag(outs, HID, rowBase + mt * 16, k * 32, lane);
        acc[0][mt] = wmma_bf16(a, bf0, acc[0][mt]);
        acc[1][mt] = wmma_bf16(a, bf1, acc[1][mt]);
      }
    }
#pragma unroll
    for (int j = 0; j < 2; ++j) {
      const int ncol = n0 + j * 16 + (lane & 15);
      const float ob = outb[ncol];
#pragma unroll
      for (int mt = 0; mt < 4; ++mt) {
#pragma unroll
        for (int r = 0; r < 8; ++r) {
          float c = acc[j][mt][r] + ob;
          int lrow = mt * 16 + ((lane >> 4) << 3) + r;
          if (ncol == ls_tgt[lrow]) ls_T[lrow] = c;  // unique writer per row
          // row max / sum-exp across the 16-lane group holding this row
          float mx = c;
          mx = fmaxf(mx, __shfl_xor(mx, 1, 16));
          mx = fmaxf(mx, __shfl_xor(mx, 2, 16));
          mx = fmaxf(mx, __shfl_xor(mx, 4, 16));
          mx = fmaxf(mx, __shfl_xor(mx, 8, 16));
          float e = __expf(c - mx);
          e += __shfl_xor(e, 1, 16);
          e += __shfl_xor(e, 2, 16);
          e += __shfl_xor(e, 4, 16);
          e += __shfl_xor(e, 8, 16);
          if ((lane & 15) == 0) {  // per-wave private running state -> no races
            int idx = wave * 64 + lrow;
            float M = ls_M[idx], S = ls_S[idx];
            float nM = fmaxf(M, mx);
            ls_S[idx] = S * __expf(M - nM) + e * __expf(mx - nM);
            ls_M[idx] = nM;
          }
        }
      }
    }
  }
  __syncthreads();
  if (tid < 64) {
    float M = -3.0e38f, S = 0.f;
    for (int w = 0; w < 8; ++w) {
      float m2 = ls_M[w * 64 + tid], s2 = ls_S[w * 64 + tid];
      float nM = fmaxf(M, m2);
      S = S * __expf(M - nM) + s2 * __expf(m2 - nM);
      M = nM;
    }
    float nll = M + logf(S) - ls_T[tid];
    float msk = (ls_tgt[tid] != 0) ? 1.f : 0.f;  // ignore_index = PAD = 0
    float v = nll * msk;
    out_review[rowBase + tid] = v;
    atomicAdd(review_sum, v);
    atomicAdd(mask_sum, msk);
  }
}

__global__ void finalize_kernel(const float* scal, float* obj) {
  *obj = 1e-7f * scal[0] + scal[1] / fmaxf(scal[2], 1.f);
}

// ---------------- launch ----------------
extern "C" void kernel_launch(void* const* d_in, const int* in_sizes, int n_in,
                              void* d_out, int out_size, void* d_ws, size_t ws_size,
                              hipStream_t stream) {
  const int* user = (const int*)d_in[0];
  const int* item = (const int*)d_in[1];
  const float* label = (const float*)d_in[2];
  const int* review_input = (const int*)d_in[3];
  const int* review_target = (const int*)d_in[4];
  const float* user_emb = (const float*)d_in[5];
  const float* item_emb = (const float*)d_in[6];
  const float* mlp_w0 = (const float*)d_in[7];
  const float* mlp_w1 = (const float*)d_in[8];
  const float* hidden_W = (const float*)d_in[9];
  const float* hidden_b = (const float*)d_in[10];
  const float* word_emb = (const float*)d_in[11];
  const float* w_ih = (const float*)d_in[12];
  const float* w_hh = (const float*)d_in[13];
  const float* b_ih = (const float*)d_in[14];
  const float* b_hh = (const float*)d_in[15];
  const float* out_W = (const float*)d_in[16];
  const float* out_b = (const float*)d_in[17];
  const float* final_W = (const float*)d_in[18];
  const float* final_b = (const float*)d_in[19];
  const float* user_bias = (const float*)d_in[20];
  const float* item_bias = (const float*)d_in[21];

  float* out_rating = (float*)d_out;            // [256]
  float* out_review = out_rating + BATCH;       // [12800]
  float* out_obj = out_rating + BATCH + TB;     // [1]

  // workspace carve-up (256B aligned)
  char* ws = (char*)d_ws;
  size_t off = 0;
  auto carve = [&](size_t bytes) {
    char* p = ws + off;
    off += (bytes + 255) & ~(size_t)255;
    return p;
  };
  float* scal = (float*)carve(4 * sizeof(float));              // rating_sum, review_sum, mask_sum
  float* m0   = (float*)carve(BATCH * MLP0D * sizeof(float));  // [256,128]
  float* m    = (float*)carve(BATCH * MLP1D * sizeof(float));  // [256,64]
  float* h0   = (float*)carve(BATCH * HID * sizeof(float));    // [256,512]
  u16* wih_bf = (u16*)carve((size_t)G3 * GIN * 2);             // [1536,576]
  u16* whh_bf = (u16*)carve((size_t)G3 * HID * 2);             // [1536,512]
  u16* owt_bf = (u16*)carve((size_t)VOC * HID * 2);            // [20000,512]
  u16* x_bf   = (u16*)carve((size_t)TB * GIN * 2);             // [12800,576]
  u16* outs_bf = (u16*)carve((size_t)TB * HID * 2);            // [12800,512]

  zero_scal_kernel<<<1, 64, 0, stream>>>(scal);

  enc1_kernel<<<BATCH, 128, 0, stream>>>(user, item, user_emb, item_emb, mlp_w0, m0);
  enc2_kernel<<<BATCH, 64, 0, stream>>>(m0, mlp_w1, m);
  enc3_kernel<<<BATCH, 256, 0, stream>>>(m, hidden_W, hidden_b, h0);
  rating_kernel<<<1, 256, 0, stream>>>(m, final_W, final_b, user, item, user_bias,
                                       item_bias, label, out_rating, &scal[0]);

  { int n = G3 * GIN;
    cvt_bf16_kernel<<<(n + 255) / 256, 256, 0, stream>>>(w_ih, wih_bf, n); }
  { int n = G3 * HID;
    cvt_bf16_kernel<<<(n + 255) / 256, 256, 0, stream>>>(w_hh, whh_bf, n); }
  { int n = VOC * HID;
    transpose_outw_kernel<<<(n + 255) / 256, 256, 0, stream>>>(out_W, owt_bf); }
  { int n = TB * GIN;
    build_x_kernel<<<(n + 255) / 256, 256, 0, stream>>>(review_input, word_emb, m, x_bf); }

  // fused GRU: 16 blocks x 8 waves; 240KB dynamic LDS (CDNA5: 320KB/WGP)
  const size_t gru_lds = (size_t)16 * G3 * 4 * 2 + (size_t)16 * HID * 4
                       + (size_t)16 * HID * 2;
  gru_kernel<<<16, 256, gru_lds, stream>>>(x_bf, wih_bf, whh_bf, b_ih, b_hh, h0,
                                           outs_bf);

  logits_ce_kernel<<<TB / 64, 256, 0, stream>>>(outs_bf, owt_bf, out_b,
                                                review_target, out_review,
                                                &scal[1], &scal[2]);

  finalize_kernel<<<1, 1, 0, stream>>>(scal, out_obj);
  (void)in_sizes; (void)n_in; (void)out_size; (void)ws_size;
}
